// RelativeMHA_40819369181576
// MI455X (gfx1250) — matrix operand order repaired and checked by reading, not compiled, not run on hardware
//
#include <hip/hip_runtime.h>
#include <hip/hip_bf16.h>

// Transformer-XL relative MHA for gfx1250 (MI455X), bf16 WMMA pipeline.
// v_wmma_f32_16x16x32_bf16 for all matmuls, flash-style online softmax,
// rel-shift resolved analytically against a per-tile LDS pos-score band,
// async global->LDS staging (ASYNCcnt) for the GEMM tiles when available.

#define BB 8
#define TT 1024
#define DD 512
#define HH 8
#define DH 64

typedef __bf16 bf16_t;
typedef bf16_t v16bf __attribute__((ext_vector_type(16)));
typedef bf16_t v8bf  __attribute__((ext_vector_type(8)));
typedef bf16_t v4bf  __attribute__((ext_vector_type(4)));
typedef float  v8f   __attribute__((ext_vector_type(8)));

#if __has_builtin(__builtin_amdgcn_global_load_async_to_lds_b128)
#define HAVE_ASYNC_LDS 1
// The builtin takes pointers to 16-byte int vectors (v4i) in AS1/AS3.
typedef int b128i __attribute__((vector_size(16)));
typedef __attribute__((address_space(1))) b128i gb128;
typedef __attribute__((address_space(3))) b128i lb128;
#endif

// Copy 16 bytes global(bf16) -> LDS. Async path uses the CDNA5
// GLOBAL_LOAD_ASYNC_TO_LDS_B128 (tracked by ASYNCcnt, no VGPR round-trip).
static __device__ __forceinline__ void stage16(const bf16_t* g, bf16_t* l) {
#ifdef HAVE_ASYNC_LDS
  // Generic->AS1: global generic addresses are the 64-bit global address.
  // Generic->AS3: low 32 bits of a generic LDS address are the LDS byte offset.
  __builtin_amdgcn_global_load_async_to_lds_b128(
      (gb128*)(uintptr_t)g, (lb128*)(uint32_t)(uintptr_t)l, 0, 0);
#else
  *(v8bf*)l = *(const v8bf*)g;
#endif
}

static __device__ __forceinline__ void stage_commit() {
#ifdef HAVE_ASYNC_LDS
#if __has_builtin(__builtin_amdgcn_s_wait_asynccnt)
  __builtin_amdgcn_s_wait_asynccnt(0);
#else
  asm volatile("s_wait_asynccnt 0" ::: "memory");
#endif
#endif
  __syncthreads();
}

static __device__ __forceinline__ v16bf frag16(const bf16_t* p0, const bf16_t* p1) {
  v8bf lo = *(const v8bf*)p0;
  v8bf hi = *(const v8bf*)p1;
  v16bf r;
#pragma unroll
  for (int i = 0; i < 8; ++i) { r[i] = lo[i]; r[i + 8] = hi[i]; }
  return r;
}

// A-fragment (16x32, rowptr = this lane's row, 32-K window):
// lane half h: elements 0..7 -> K = 8h.. ; elements 8..15 -> K = 16+8h..
static __device__ __forceinline__ v16bf load_a_frag(const bf16_t* rowptr, int half) {
  return frag16(rowptr + 8 * half, rowptr + 16 + 8 * half);
}
// B-fragment (32x16, colptr = this lane's column base, K contiguous):
// lane half h: elements 0..7 -> K = 8h.. ; elements 8..15 -> K = 16+8h..
static __device__ __forceinline__ v16bf load_b_frag(const bf16_t* colptr, int half) {
  return frag16(colptr + 8 * half, colptr + 16 + 8 * half);
}

static __device__ __forceinline__ v8f wmma_bf16(v16bf a, v16bf b, v8f c) {
  // (neg_a, A, neg_b, B, c_mod, C, reuse_a, reuse_b)
  return __builtin_amdgcn_wmma_f32_16x16x32_bf16(false, a, false, b, (short)0, c,
                                                 false, false);
}

// ---------------------------------------------------------------------------
// Kernel 0a: one-shot f32 -> bf16 conversion of the 4 activation tensors.
// ---------------------------------------------------------------------------
__global__ __launch_bounds__(256)
void convert_x_kernel(const float* __restrict__ q_in, const float* __restrict__ k_in,
                      const float* __restrict__ v_in, const float* __restrict__ p_in,
                      bf16_t* __restrict__ xq, bf16_t* __restrict__ xk,
                      bf16_t* __restrict__ xv, bf16_t* __restrict__ xp) {
  const float* src;
  bf16_t* dst;
  switch (blockIdx.y) {
    case 0:  src = q_in; dst = xq; break;
    case 1:  src = k_in; dst = xk; break;
    case 2:  src = v_in; dst = xv; break;
    default: src = p_in; dst = xp; break;
  }
  size_t i0 = ((size_t)blockIdx.x * 256 + threadIdx.x) * 4;
  float4 v = *(const float4*)(src + i0);
  v4bf o;
  o[0] = (bf16_t)v.x; o[1] = (bf16_t)v.y; o[2] = (bf16_t)v.z; o[3] = (bf16_t)v.w;
  *(v4bf*)(dst + i0) = o;
}

// ---------------------------------------------------------------------------
// Kernel 0b: weights f32 -> bf16, transposed to [n][k] panels so B-fragment
// K-runs are contiguous and tiles can be async-copied straight into LDS.
// wt layout: [5][DD cols(n)][DD k], order Wq,Wk,Wv,Wp,Wo.
// ---------------------------------------------------------------------------
__global__ __launch_bounds__(256)
void convert_w_kernel(const float* __restrict__ Wq, const float* __restrict__ Wk,
                      const float* __restrict__ Wv, const float* __restrict__ Wp,
                      const float* __restrict__ Wo, bf16_t* __restrict__ wt) {
  const float* W;
  switch (blockIdx.y) {
    case 0:  W = Wq; break;
    case 1:  W = Wk; break;
    case 2:  W = Wv; break;
    case 3:  W = Wp; break;
    default: W = Wo; break;
  }
  bf16_t* dst = wt + (size_t)blockIdx.y * DD * DD;
  int idx = blockIdx.x * 256 + threadIdx.x;
  int n = idx & (DD - 1), k = idx >> 9;
  dst[(size_t)n * DD + k] = (bf16_t)W[(size_t)k * DD + n];  // coalesced read over n
}

// ---------------------------------------------------------------------------
// Kernel 1: input projections. 64x64 output tile / WG, async-staged LDS tiles.
// proj: 0=q (writes qu=q+u_bias and qvb=q+v_bias), 1=k, 2=v (transposed), 3=p
// ---------------------------------------------------------------------------
__global__ __launch_bounds__(256)
void proj_kernel(const bf16_t* __restrict__ xq, const bf16_t* __restrict__ xk,
                 const bf16_t* __restrict__ xv, const bf16_t* __restrict__ xp,
                 const bf16_t* __restrict__ wt,
                 const float* __restrict__ bq, const float* __restrict__ bk,
                 const float* __restrict__ bv, const float* __restrict__ bp,
                 const float* __restrict__ ub, const float* __restrict__ vb,
                 bf16_t* __restrict__ qu, bf16_t* __restrict__ qvb,
                 bf16_t* __restrict__ ko, bf16_t* __restrict__ vt,
                 bf16_t* __restrict__ po) {
  const int proj = blockIdx.z;
  const bf16_t* X;
  const float* bias;
  if (proj == 0)      { X = xq; bias = bq; }
  else if (proj == 1) { X = xk; bias = bk; }
  else if (proj == 2) { X = xv; bias = bv; }
  else                { X = xp; bias = bp; }
  const bf16_t* W = wt + (size_t)proj * DD * DD;  // [n][k]

  const int colBase = blockIdx.x * 64;
  const int rowBase = blockIdx.y * 64;
  const int tid = threadIdx.x;
  const int wave = tid >> 5, lane = tid & 31, half = lane >> 4, l16 = lane & 15;
  const int m0 = (wave & 3) * 16;   // 4 row-bands
  const int n0 = (wave >> 2) * 32;  // 2 col-halves, 2 subtiles each

  __shared__ __attribute__((aligned(16))) bf16_t Xs[64 * 40]; // [row][k], pad 40
  __shared__ __attribute__((aligned(16))) bf16_t Ws[64 * 40]; // [n][k]

  const int srow = tid >> 2;          // 0..63: row/col of the staged tile
  const int schunk = (tid & 3) * 8;   // 16B chunk within the 32-K window

  v8f acc0 = {}, acc1 = {};

  for (int kk = 0; kk < DD; kk += 32) {
    __syncthreads();  // previous iteration's fragment reads complete
    stage16(X + (size_t)(rowBase + srow) * DD + kk + schunk, &Xs[srow * 40 + schunk]);
    stage16(W + (size_t)(colBase + srow) * DD + kk + schunk, &Ws[srow * 40 + schunk]);
    stage_commit();
    v16bf a  = load_a_frag(&Xs[(m0 + l16) * 40], half);
    v16bf b0 = load_b_frag(&Ws[(n0 + l16) * 40], half);
    v16bf b1 = load_b_frag(&Ws[(n0 + 16 + l16) * 40], half);
    acc0 = wmma_bf16(a, b0, acc0);
    acc1 = wmma_bf16(a, b1, acc1);
  }

#pragma unroll
  for (int s = 0; s < 2; ++s) {
    v8f acc = s ? acc1 : acc0;
#pragma unroll
    for (int r = 0; r < 8; ++r) {
      int m = m0 + r + 8 * half;           // C layout: row = r + 8*(lane/16)
      int grow = rowBase + m;
      int bidx = grow >> 10;               // / T
      int tok = grow & (TT - 1);
      int gcol = colBase + n0 + s * 16 + l16;
      int h = gcol >> 6, d = gcol & 63;
      float val = acc[r] + bias[gcol];
      size_t oh = (((size_t)bidx * HH + h) * TT + tok) * DH + d;
      if (proj == 0) {
        qu[oh]  = (bf16_t)(val + ub[gcol]);
        qvb[oh] = (bf16_t)(val + vb[gcol]);
      } else if (proj == 1) {
        ko[oh] = (bf16_t)val;
      } else if (proj == 2) {
        vt[(((size_t)bidx * HH + h) * DH + d) * TT + tok] = (bf16_t)val; // transposed
      } else {
        po[oh] = (bf16_t)val;
      }
    }
  }
}

// ---------------------------------------------------------------------------
// Kernel 2: fused scores + rel-shift + online softmax + attn*V.
// One WG = (b, h, 16 query rows). 8 waves split key dim (128 cols each).
// ---------------------------------------------------------------------------
#define PSTR 1032

__global__ __launch_bounds__(256)
void attn_kernel(const bf16_t* __restrict__ qu, const bf16_t* __restrict__ qvb,
                 const bf16_t* __restrict__ ko, const bf16_t* __restrict__ vt,
                 const bf16_t* __restrict__ po, bf16_t* __restrict__ ctx) {
  const int r0 = blockIdx.x * 16;
  const int h  = blockIdx.y;
  const int b  = blockIdx.z;
  const int tid = threadIdx.x;
  const int wave = tid >> 5, lane = tid & 31, half = lane >> 4, l16 = lane & 15;

  __shared__ __attribute__((aligned(16))) bf16_t Pl[17 * PSTR];     // pos rows r0..r0+16
  __shared__ __attribute__((aligned(16))) bf16_t Es[8][16 * 32];    // per-wave exp(S)
  __shared__ __attribute__((aligned(16))) bf16_t Cp[8][16][64];     // ctx partials
  __shared__ float2 St[8][16];                                      // (m, l) partials

  const size_t bh = (size_t)b * HH + h;
  const bf16_t* qvRow = qvb + (bh * TT + r0 + l16) * DH;
  const bf16_t* quRow = qu  + (bh * TT + r0 + l16) * DH;

  // ---- Phase 1: P = (q+v_bias) @ p^T for 16 rows x all T cols, into LDS ----
  v16bf aqv0 = load_a_frag(qvRow + 0, half);
  v16bf aqv1 = load_a_frag(qvRow + 32, half);
#pragma unroll
  for (int c = 0; c < 8; ++c) {
    int colBase = wave * 128 + c * 16;
    const bf16_t* pcol = po + (bh * TT + colBase + l16) * DH;
    v8f acc = {};
    acc = wmma_bf16(aqv0, load_b_frag(pcol + 0, half), acc);
    acc = wmma_bf16(aqv1, load_b_frag(pcol + 32, half), acc);
#pragma unroll
    for (int r = 0; r < 8; ++r)
      Pl[(r + 8 * half) * PSTR + colBase + l16] = (bf16_t)acc[r];
  }
  { // wrap row r0+16 (used by shifted row r0+15 for j' > i'+1), scalar dots
    int r16 = r0 + 16;
#pragma unroll
    for (int j = 0; j < 4; ++j) {
      int c = tid * 4 + j;
      float s = 0.f;
      if (r16 < TT) {
        const bf16_t* qr = qvb + (bh * TT + r16) * DH;
        const bf16_t* pr = po + (bh * TT + c) * DH;
        for (int d = 0; d < DH; ++d) s += (float)qr[d] * (float)pr[d];
      }
      Pl[16 * PSTR + c] = (bf16_t)s;
    }
  }
  __syncthreads();

  // ---- Phase 2: streaming scores + online softmax over wave's 128 cols ----
  v16bf aqu0 = load_a_frag(quRow + 0, half);
  v16bf aqu1 = load_a_frag(quRow + 32, half);

  float m[8], l[8];
  v8f cacc[4];
  v8f vzero = {};
#pragma unroll
  for (int r = 0; r < 8; ++r) { m[r] = -__builtin_inff(); l[r] = 0.f; }
#pragma unroll
  for (int dt = 0; dt < 4; ++dt) cacc[dt] = vzero;

  const float rs = 0.04419417382415922f; // 1/sqrt(D)

  for (int kb = 0; kb < 4; ++kb) {
    int c0 = wave * 128 + kb * 32;
    if (kb < 3) {  // prime next K/V block (global_prefetch_b8)
      __builtin_prefetch(ko + (bh * TT + c0 + 32 + l16) * DH, 0, 1);
      __builtin_prefetch(vt + (bh * DH + l16) * TT + c0 + 32, 0, 1);
    }
    float sc[2][8];
#pragma unroll
    for (int s = 0; s < 2; ++s) {
      const bf16_t* kcol = ko + (bh * TT + c0 + 16 * s + l16) * DH;
      v8f acc = {};
      acc = wmma_bf16(aqu0, load_b_frag(kcol + 0, half), acc);
      acc = wmma_bf16(aqu1, load_b_frag(kcol + 32, half), acc);
#pragma unroll
      for (int r = 0; r < 8; ++r) {
        int M = r + 8 * half;
        int ip = r0 + M;             // i' (query row)
        int jp = c0 + 16 * s + l16;  // j' (key col)
        float pos;
        if (jp <= ip)           pos = (float)Pl[M * PSTR + (TT - 1 - ip + jp)];
        else if (jp == ip + 1)  pos = 0.f;
        else                    pos = (float)Pl[(M + 1) * PSTR + (jp - ip - 2)];
        sc[s][r] = (acc[r] + pos) * rs;
      }
    }
    // row-max across 32 cols (rows live within 16-lane halves; wave32 shfl)
    float rmax[8];
#pragma unroll
    for (int r = 0; r < 8; ++r) rmax[r] = fmaxf(sc[0][r], sc[1][r]);
#pragma unroll
    for (int mask = 1; mask < 16; mask <<= 1)
#pragma unroll
      for (int r = 0; r < 8; ++r)
        rmax[r] = fmaxf(rmax[r], __shfl_xor(rmax[r], mask, 32));
    float scale[8], rsum[8];
#pragma unroll
    for (int r = 0; r < 8; ++r) {
      float mn = fmaxf(m[r], rmax[r]);
      scale[r] = __expf(m[r] - mn);
      m[r] = mn;
      rsum[r] = 0.f;
    }
#pragma unroll
    for (int s = 0; s < 2; ++s)
#pragma unroll
      for (int r = 0; r < 8; ++r) {
        float e = __expf(sc[s][r] - m[r]);
        rsum[r] += e;
        Es[wave][(r + 8 * half) * 32 + s * 16 + l16] = (bf16_t)e; // C -> A relayout
      }
#pragma unroll
    for (int mask = 1; mask < 16; mask <<= 1)
#pragma unroll
      for (int r = 0; r < 8; ++r)
        rsum[r] += __shfl_xor(rsum[r], mask, 32);
#pragma unroll
    for (int r = 0; r < 8; ++r) l[r] = l[r] * scale[r] + rsum[r];
#pragma unroll
    for (int dt = 0; dt < 4; ++dt)
#pragma unroll
      for (int r = 0; r < 8; ++r) cacc[dt][r] *= scale[r];
    // ctx += exp(S) @ V  (A from per-wave LDS; B contiguous from transposed V)
    v16bf ae = load_a_frag(&Es[wave][l16 * 32], half);
#pragma unroll
    for (int dt = 0; dt < 4; ++dt) {
      const bf16_t* vcol = vt + (bh * DH + dt * 16 + l16) * TT + c0;
      cacc[dt] = wmma_bf16(ae, load_b_frag(vcol, half), cacc[dt]);
    }
  }

  // ---- write per-wave partials, then cross-wave combine ----
#pragma unroll
  for (int dt = 0; dt < 4; ++dt)
#pragma unroll
    for (int r = 0; r < 8; ++r)
      Cp[wave][r + 8 * half][dt * 16 + l16] = (bf16_t)cacc[dt][r];
  if (l16 == 0)
#pragma unroll
    for (int r = 0; r < 8; ++r)
      St[wave][r + 8 * half] = make_float2(m[r], l[r]);
  __syncthreads();

#pragma unroll
  for (int j = 0; j < 4; ++j) {
    int e = tid + 256 * j;
    int M = e >> 6, d = e & 63;
    float gm = -__builtin_inff();
#pragma unroll
    for (int w = 0; w < 8; ++w) gm = fmaxf(gm, St[w][M].x);
    float Lsum = 0.f, num = 0.f;
#pragma unroll
    for (int w = 0; w < 8; ++w) {
      float f = __expf(St[w][M].x - gm);
      Lsum += St[w][M].y * f;
      num += (float)Cp[w][M][d] * f;
    }
    ctx[((size_t)b * TT + r0 + M) * DD + h * DH + d] = (bf16_t)(num / Lsum);
  }
}

// ---------------------------------------------------------------------------
// Kernel 3: output projection: out = ctx(bf16) @ Wo + bo  (f32 out)
// ---------------------------------------------------------------------------
__global__ __launch_bounds__(256)
void out_proj_kernel(const bf16_t* __restrict__ ctx, const bf16_t* __restrict__ W,
                     const float* __restrict__ bo, float* __restrict__ out) {
  const int colBase = blockIdx.x * 64;
  const int rowBase = blockIdx.y * 64;
  const int tid = threadIdx.x;
  const int wave = tid >> 5, lane = tid & 31, half = lane >> 4, l16 = lane & 15;
  const int m0 = (wave & 3) * 16;
  const int n0 = (wave >> 2) * 32;

  __shared__ __attribute__((aligned(16))) bf16_t Xs[64 * 40];
  __shared__ __attribute__((aligned(16))) bf16_t Ws[64 * 40];

  const int srow = tid >> 2;
  const int schunk = (tid & 3) * 8;

  v8f acc0 = {}, acc1 = {};

  for (int kk = 0; kk < DD; kk += 32) {
    __syncthreads();
    stage16(ctx + (size_t)(rowBase + srow) * DD + kk + schunk, &Xs[srow * 40 + schunk]);
    stage16(W + (size_t)(colBase + srow) * DD + kk + schunk, &Ws[srow * 40 + schunk]);
    stage_commit();
    v16bf a  = load_a_frag(&Xs[(m0 + l16) * 40], half);
    v16bf b0 = load_b_frag(&Ws[(n0 + l16) * 40], half);
    v16bf b1 = load_b_frag(&Ws[(n0 + 16 + l16) * 40], half);
    acc0 = wmma_bf16(a, b0, acc0);
    acc1 = wmma_bf16(a, b1, acc1);
  }

#pragma unroll
  for (int s = 0; s < 2; ++s) {
    v8f acc = s ? acc1 : acc0;
#pragma unroll
    for (int r = 0; r < 8; ++r) {
      int m = m0 + r + 8 * half;
      int gcol = colBase + n0 + s * 16 + l16;
      out[(size_t)(rowBase + m) * DD + gcol] = acc[r] + bo[gcol];
    }
  }
}

// ---------------------------------------------------------------------------
extern "C" void kernel_launch(void* const* d_in, const int* in_sizes, int n_in,
                              void* d_out, int out_size, void* d_ws, size_t ws_size,
                              hipStream_t stream) {
  (void)in_sizes; (void)n_in; (void)out_size; (void)ws_size;
  const float* q_in = (const float*)d_in[0];
  const float* k_in = (const float*)d_in[1];
  const float* v_in = (const float*)d_in[2];
  const float* p_in = (const float*)d_in[3];
  const float* Wq = (const float*)d_in[4];
  const float* bq = (const float*)d_in[5];
  const float* Wk = (const float*)d_in[6];
  const float* bk = (const float*)d_in[7];
  const float* Wv = (const float*)d_in[8];
  const float* bv = (const float*)d_in[9];
  const float* Wp = (const float*)d_in[10];
  const float* bp = (const float*)d_in[11];
  const float* Wo = (const float*)d_in[12];
  const float* bo = (const float*)d_in[13];
  const float* ub = (const float*)d_in[14];
  const float* vb = (const float*)d_in[15];

  const size_t NE = (size_t)BB * TT * DD;  // 4M elements per [B,T,D] bf16 buffer
  bf16_t* qu  = (bf16_t*)d_ws;
  bf16_t* qvb = qu + NE;
  bf16_t* ko  = qvb + NE;
  bf16_t* vt  = ko + NE;
  bf16_t* po  = vt + NE;
  bf16_t* ctx = po + NE;
  bf16_t* xq  = ctx + NE;
  bf16_t* xk  = xq + NE;
  bf16_t* xv  = xk + NE;
  bf16_t* xp  = xv + NE;
  bf16_t* wtb = xp + NE;  // 5 * DD*DD transposed bf16 weights (~87 MB total ws)

  convert_x_kernel<<<dim3(NE / 1024, 4), 256, 0, stream>>>(
      q_in, k_in, v_in, p_in, xq, xk, xv, xp);
  convert_w_kernel<<<dim3((DD * DD) / 256, 5), 256, 0, stream>>>(
      Wq, Wk, Wv, Wp, Wo, wtb);
  proj_kernel<<<dim3(DD / 64, (BB * TT) / 64, 4), 256, 0, stream>>>(
      xq, xk, xv, xp, wtb, bq, bk, bv, bp, ub, vb, qu, qvb, ko, vt, po);
  attn_kernel<<<dim3(TT / 16, HH, BB), 256, 0, stream>>>(qu, qvb, ko, vt, po, ctx);
  out_proj_kernel<<<dim3(DD / 64, (BB * TT) / 64), 256, 0, stream>>>(
      ctx, wtb + (size_t)4 * DD * DD, bo, (float*)d_out);
}
